// CausalSelfAttention_14903536517156
// MI455X (gfx1250) — matrix operand order repaired and checked by reading, not compiled, not run on hardware
//
#include <hip/hip_runtime.h>
#include <hip/hip_bf16.h>

typedef __attribute__((ext_vector_type(16))) _Float16 v16h;
typedef __attribute__((ext_vector_type(8)))  _Float16 v8h;
typedef __attribute__((ext_vector_type(4)))  _Float16 v4h;
typedef __attribute__((ext_vector_type(8)))  float    v8f;

#define BDIM  16
#define TDIM  1024
#define CDIM  768
#define NHEAD 12
#define HDIM  64

__device__ inline v8f wmma16(v16h a, v16h b, v8f c) {
  return __builtin_amdgcn_wmma_f32_16x16x32_f16(false, a, false, b, (short)0, c,
                                                false, false);
}

__device__ inline v8f vzero8() {
  v8f z;
#pragma unroll
  for (int i = 0; i < 8; ++i) z[i] = 0.f;
  return z;
}

// Async copy of 16B per lane from global to LDS (ASYNCcnt-tracked).
// LDS byte address = low 32 bits of the generic pointer to __shared__.
__device__ inline void async_b128(unsigned lds_addr, unsigned goff_bytes,
                                  const void* sbase) {
  asm volatile("global_load_async_to_lds_b128 %0, %1, %2"
               :
               : "v"(lds_addr), "v"(goff_bytes), "s"(sbase)
               : "memory");
}
__device__ inline void wait_async0() {
  asm volatile("s_wait_asynccnt 0x0" ::: "memory");
}

// ---------------------------------------------------------------------------
// Prep kernels: f32 -> f16 convert, and weight transpose to [N][K] f16.
// ---------------------------------------------------------------------------
__global__ __launch_bounds__(256) void cvt_f16_kernel(
    const float* __restrict__ src, _Float16* __restrict__ dst, int n4) {
  int i = blockIdx.x * 256 + threadIdx.x;
  if (i < n4) {
    float4 v = ((const float4*)src)[i];
    v4h o;
    o[0] = (_Float16)v.x; o[1] = (_Float16)v.y;
    o[2] = (_Float16)v.z; o[3] = (_Float16)v.w;
    ((v4h*)dst)[i] = o;
  }
}

__global__ __launch_bounds__(256) void transpose_f16_kernel(
    const float* __restrict__ W, _Float16* __restrict__ Wt, int K, int N) {
  int idx = blockIdx.x * 256 + threadIdx.x;
  if (idx < K * N) {
    int k = idx / N, n = idx - k * N;
    Wt[(size_t)n * K + k] = (_Float16)W[idx];
  }
}

// ---------------------------------------------------------------------------
// Stage one 128x64 f16 tile (rows row0.., cols k0..k0+63 of a [rows][768]
// f16 matrix) into LDS via async b128 copies. 1024 chunks, 4 per thread.
// ---------------------------------------------------------------------------
__device__ inline void stage_tile_async(const _Float16* gbase, unsigned row0,
                                        unsigned k0, _Float16* ltile, int tid) {
#pragma unroll
  for (int i = 0; i < 4; ++i) {
    unsigned ch = (unsigned)tid * 4u + (unsigned)i;
    unsigned row = ch >> 3;
    unsigned c8 = (ch & 7u) << 3;
    unsigned goff = ((row0 + row) * 768u + k0 + c8) * 2u;
    unsigned laddr = (unsigned)(size_t)(ltile + row * 64 + c8);
    async_b128(laddr, goff, gbase);
  }
}

// ---------------------------------------------------------------------------
// Shared GEMM inner step: 16 WMMAs on a staged 128x64 A-tile / B-tile pair.
// acc layout: [4 m-subtiles][2 n-subtiles].
// ---------------------------------------------------------------------------
__device__ inline void gemm_step(const _Float16* cA, const _Float16* cB, int wm,
                                 int wn, int ln, int hid, v8f acc[4][2]) {
#pragma unroll
  for (int kc = 0; kc < 2; ++kc) {
    v16h afr[4], bfr[2];
#pragma unroll
    for (int mt = 0; mt < 4; ++mt) {
      const _Float16* ap = cA + (wm * 64 + mt * 16 + ln) * 64 + kc * 32;
      v8h lo = *(const v8h*)(ap + hid * 16);
      v8h hi = *(const v8h*)(ap + hid * 16 + 8);
      v16h f;
#pragma unroll
      for (int j = 0; j < 8; ++j) { f[j] = lo[j]; f[8 + j] = hi[j]; }
      afr[mt] = f;
    }
#pragma unroll
    for (int nt = 0; nt < 2; ++nt)
      bfr[nt] = *(const v16h*)(cB + (wn * 32 + nt * 16 + ln) * 64 + kc * 32 +
                               hid * 16);
#pragma unroll
    for (int mt = 0; mt < 4; ++mt)
#pragma unroll
      for (int nt = 0; nt < 2; ++nt)
        acc[mt][nt] = wmma16(afr[mt], bfr[nt], acc[mt][nt]);
  }
}

// ---------------------------------------------------------------------------
// Kernel 1: qkv = Xh(f16) @ WTattn^T + b_attn   (M=16384, N=2304, K=768)
// Epilogue scatters to head-major f16 Q (pre-scaled 1/8), K, V.
// ---------------------------------------------------------------------------
__global__ __launch_bounds__(256) void qkv_wmma_kernel(
    const _Float16* __restrict__ Ah, const _Float16* __restrict__ Bt,
    const float* __restrict__ bias, _Float16* __restrict__ Qb,
    _Float16* __restrict__ Kb, _Float16* __restrict__ Vb) {
  __shared__ __align__(32) _Float16 sA[2][128 * 64];
  __shared__ __align__(32) _Float16 sB[2][128 * 64];

  const int tid  = threadIdx.x;
  const int lane = tid & 31;
  const int wave = tid >> 5;
  const int wm   = wave >> 2;   // 0..1
  const int wn   = wave & 3;    // 0..3
  const int ln   = lane & 15;
  const int hid  = lane >> 4;
  const int m0   = blockIdx.y * 128;
  const int n0   = blockIdx.x * 128;

  v8f acc[4][2];
#pragma unroll
  for (int mt = 0; mt < 4; ++mt)
#pragma unroll
    for (int nt = 0; nt < 2; ++nt) acc[mt][nt] = vzero8();

  stage_tile_async(Ah, m0, 0, &sA[0][0], tid);
  stage_tile_async(Bt, n0, 0, &sB[0][0], tid);

  for (int kt = 0; kt < 12; ++kt) {
    const int cur = kt & 1;
    wait_async0();
    __syncthreads();
    if (kt + 1 < 12) {
      stage_tile_async(Ah, m0, (unsigned)(kt + 1) * 64u, &sA[1 - cur][0], tid);
      stage_tile_async(Bt, n0, (unsigned)(kt + 1) * 64u, &sB[1 - cur][0], tid);
    }
    gemm_step(&sA[cur][0], &sB[cur][0], wm, wn, ln, hid, acc);
  }

#pragma unroll
  for (int mt = 0; mt < 4; ++mt) {
#pragma unroll
    for (int nt = 0; nt < 2; ++nt) {
      const int n = n0 + wn * 32 + nt * 16 + ln;
      const int which = n / CDIM;     // 0=q 1=k 2=v
      const int c = n - which * CDIM;
      const int hh = c >> 6, dd = c & 63;
      const float bv = bias[n];
      _Float16* dst = (which == 0) ? Qb : ((which == 1) ? Kb : Vb);
      const float scale = (which == 0) ? 0.125f : 1.0f;
#pragma unroll
      for (int i = 0; i < 8; ++i) {
        const int m = m0 + wm * 64 + mt * 16 + i + 8 * hid;
        const int bidx = m >> 10, trow = m & 1023;
        const float v = (acc[mt][nt][i] + bv) * scale;
        dst[(((size_t)(bidx * NHEAD + hh)) * TDIM + trow) * HDIM + dd] =
            (_Float16)v;
      }
    }
  }
}

// ---------------------------------------------------------------------------
// Kernel 2: causal flash attention; cooperative K/V LDS staging.
// Block = (128 queries, head, batch); 8 waves x 16 query rows.
// ---------------------------------------------------------------------------
__global__ __launch_bounds__(256) void flash_attn_kernel(
    const _Float16* __restrict__ Q, const _Float16* __restrict__ K,
    const _Float16* __restrict__ V, _Float16* __restrict__ O) {
  __shared__ __align__(32) _Float16 sK[32 * 64];   // [key][dim]
  __shared__ __align__(32) _Float16 sVt[64 * 32];  // [dim][key] (transposed)
  __shared__ __align__(32) _Float16 sP[8][16 * 32];

  const int tid = threadIdx.x;
  const int lane = tid & 31;
  const int wave = tid >> 5;
  const int ln = lane & 15, hid = lane >> 4;
  const int h = blockIdx.y, b = blockIdx.z;
  const int qbase = blockIdx.x * 128;
  const int q0 = qbase + wave * 16;

  const size_t hoff = ((size_t)(b * NHEAD + h)) * TDIM * HDIM;
  const _Float16* Qh = Q + hoff;
  const _Float16* Kh = K + hoff;
  const _Float16* Vh = V + hoff;

  v16h qf[2];
  {
    const _Float16* qp = Qh + (size_t)(q0 + ln) * HDIM;
#pragma unroll
    for (int s = 0; s < 2; ++s) {
      v8h lo = *(const v8h*)(qp + s * 32 + hid * 16);
      v8h hi = *(const v8h*)(qp + s * 32 + hid * 16 + 8);
      v16h f;
#pragma unroll
      for (int j = 0; j < 8; ++j) { f[j] = lo[j]; f[8 + j] = hi[j]; }
      qf[s] = f;
    }
  }

  v8f acc[4];
#pragma unroll
  for (int t2 = 0; t2 < 4; ++t2) acc[t2] = vzero8();
  float rmax[8], rsum[8];
#pragma unroll
  for (int i = 0; i < 8; ++i) { rmax[i] = -3.0e38f; rsum[i] = 0.f; }

  const int nblk = (qbase + 128) >> 5;  // uniform across the workgroup
  const unsigned row = (unsigned)tid >> 3;        // staging chunk decode
  const unsigned c8 = ((unsigned)tid & 7u) << 3;

  for (int blk = 0; blk < nblk; ++blk) {
    const int kb0 = blk << 5;
    __syncthreads();  // previous tile fully consumed
    // K tile: async raw copy [32][64] f16 (one b128 per thread)
    async_b128((unsigned)(size_t)(sK + row * 64 + c8),
               (((unsigned)kb0 + row) * 64u + c8) * 2u, Kh);
    // V tile: manual transposed staging -> sVt[dim][key]
    {
      v8h hv = *(const v8h*)(Vh + (size_t)(kb0 + row) * HDIM + c8);
#pragma unroll
      for (int e = 0; e < 8; ++e) sVt[(c8 + e) * 32 + row] = hv[e];
    }
    wait_async0();
    __syncthreads();

    if (kb0 <= q0 + 15) {  // wave-uniform causal skip; EXEC stays all-1s
      v8f sc[2];
#pragma unroll
      for (int t = 0; t < 2; ++t) {
        v8f st = vzero8();
#pragma unroll
        for (int s = 0; s < 2; ++s) {
          v16h kf =
              *(const v16h*)(sK + (t * 16 + ln) * 64 + s * 32 + hid * 16);
          st = wmma16(qf[s], kf, st);
        }
        sc[t] = st;
      }
      float alpha[8];
#pragma unroll
      for (int i = 0; i < 8; ++i) {
        const int rowq = q0 + i + 8 * hid;
        float a0 = (kb0 + ln      <= rowq) ? sc[0][i] : -3.0e38f;
        float a1 = (kb0 + 16 + ln <= rowq) ? sc[1][i] : -3.0e38f;
        float v = fmaxf(a0, a1);
        v = fmaxf(v, __shfl_xor(v, 1));
        v = fmaxf(v, __shfl_xor(v, 2));
        v = fmaxf(v, __shfl_xor(v, 4));
        v = fmaxf(v, __shfl_xor(v, 8));
        const float nm = fmaxf(rmax[i], v);
        alpha[i] = __expf(rmax[i] - nm);
        rmax[i] = nm;
        const float p0 = __expf(a0 - nm);
        const float p1 = __expf(a1 - nm);
        sc[0][i] = p0; sc[1][i] = p1;
        float ps = p0 + p1;
        ps += __shfl_xor(ps, 1);
        ps += __shfl_xor(ps, 2);
        ps += __shfl_xor(ps, 4);
        ps += __shfl_xor(ps, 8);
        rsum[i] = rsum[i] * alpha[i] + ps;
      }
      // C-layout -> A-layout transpose of P via per-wave LDS tile
      _Float16* pw = &sP[wave][0];
#pragma unroll
      for (int t = 0; t < 2; ++t)
#pragma unroll
        for (int i = 0; i < 8; ++i)
          pw[(i + 8 * hid) * 32 + t * 16 + ln] = (_Float16)sc[t][i];
      asm volatile("s_wait_dscnt 0x0" ::: "memory");
      v16h pf;
      {
        const _Float16* pp = pw + ln * 32;
        v8h lo = *(const v8h*)(pp + hid * 16);
        v8h hi = *(const v8h*)(pp + hid * 16 + 8);
#pragma unroll
        for (int j = 0; j < 8; ++j) { pf[j] = lo[j]; pf[8 + j] = hi[j]; }
      }
#pragma unroll
      for (int t2 = 0; t2 < 4; ++t2) {
#pragma unroll
        for (int i = 0; i < 8; ++i) acc[t2][i] *= alpha[i];
        v16h vf = *(const v16h*)(sVt + (t2 * 16 + ln) * 32 + hid * 16);
        acc[t2] = wmma16(pf, vf, acc[t2]);
      }
    }
  }

#pragma unroll
  for (int i = 0; i < 8; ++i) {
    const float inv = 1.0f / rsum[i];
    const int rowq = q0 + i + 8 * hid;
#pragma unroll
    for (int t2 = 0; t2 < 4; ++t2)
      O[((size_t)(b * TDIM + rowq)) * CDIM + h * HDIM + t2 * 16 + ln] =
          (_Float16)(acc[t2][i] * inv);
  }
}

// ---------------------------------------------------------------------------
// Kernel 3: out = Ab(f16) @ WTproj^T + b_proj  (M=16384, N=768, K=768), f32 out
// ---------------------------------------------------------------------------
__global__ __launch_bounds__(256) void proj_wmma_kernel(
    const _Float16* __restrict__ Ah, const _Float16* __restrict__ Bt,
    const float* __restrict__ bias, float* __restrict__ Out) {
  __shared__ __align__(32) _Float16 sA[2][128 * 64];
  __shared__ __align__(32) _Float16 sB[2][128 * 64];

  const int tid  = threadIdx.x;
  const int lane = tid & 31;
  const int wave = tid >> 5;
  const int wm   = wave >> 2;
  const int wn   = wave & 3;
  const int ln   = lane & 15;
  const int hid  = lane >> 4;
  const int m0   = blockIdx.y * 128;
  const int n0   = blockIdx.x * 128;

  v8f acc[4][2];
#pragma unroll
  for (int mt = 0; mt < 4; ++mt)
#pragma unroll
    for (int nt = 0; nt < 2; ++nt) acc[mt][nt] = vzero8();

  stage_tile_async(Ah, m0, 0, &sA[0][0], tid);
  stage_tile_async(Bt, n0, 0, &sB[0][0], tid);

  for (int kt = 0; kt < 12; ++kt) {
    const int cur = kt & 1;
    wait_async0();
    __syncthreads();
    if (kt + 1 < 12) {
      stage_tile_async(Ah, m0, (unsigned)(kt + 1) * 64u, &sA[1 - cur][0], tid);
      stage_tile_async(Bt, n0, (unsigned)(kt + 1) * 64u, &sB[1 - cur][0], tid);
    }
    gemm_step(&sA[cur][0], &sB[cur][0], wm, wn, ln, hid, acc);
  }

#pragma unroll
  for (int mt = 0; mt < 4; ++mt) {
#pragma unroll
    for (int nt = 0; nt < 2; ++nt) {
      const int n = n0 + wn * 32 + nt * 16 + ln;
      const float bv = bias[n];
#pragma unroll
      for (int i = 0; i < 8; ++i) {
        const int m = m0 + wm * 64 + mt * 16 + i + 8 * hid;
        Out[(size_t)m * CDIM + n] = acc[mt][nt][i] + bv;
      }
    }
  }
}

// ---------------------------------------------------------------------------
extern "C" void kernel_launch(void* const* d_in, const int* in_sizes, int n_in,
                              void* d_out, int out_size, void* d_ws,
                              size_t ws_size, hipStream_t stream) {
  (void)in_sizes; (void)n_in; (void)out_size; (void)ws_size;
  const float* x      = (const float*)d_in[0];
  const float* W_attn = (const float*)d_in[1];
  const float* b_attn = (const float*)d_in[2];
  const float* W_proj = (const float*)d_in[3];
  const float* b_proj = (const float*)d_in[4];
  float* out = (float*)d_out;

  const size_t HE = (size_t)BDIM * NHEAD * TDIM * HDIM;  // 12,582,912
  _Float16* Qb  = (_Float16*)d_ws;
  _Float16* Kb  = Qb + HE;
  _Float16* Vb  = Kb + HE;
  _Float16* Ab  = Vb + HE;          // attention output [B,T,C] f16
  _Float16* Xh  = Ab + HE;          // x in f16 [M,768]
  _Float16* WTa = Xh + HE;          // W_attn^T f16 [2304][768]
  _Float16* WTp = WTa + (size_t)CDIM * 3 * CDIM;  // W_proj^T f16 [768][768]

  // prep: conversions + weight transposes
  cvt_f16_kernel<<<(int)(HE / 4 + 255) / 256, 256, 0, stream>>>(x, Xh,
                                                                (int)(HE / 4));
  transpose_f16_kernel<<<(CDIM * 3 * CDIM + 255) / 256, 256, 0, stream>>>(
      W_attn, WTa, CDIM, 3 * CDIM);
  transpose_f16_kernel<<<(CDIM * CDIM + 255) / 256, 256, 0, stream>>>(
      W_proj, WTp, CDIM, CDIM);

  qkv_wmma_kernel<<<dim3((3 * CDIM) / 128, (BDIM * TDIM) / 128), 256, 0,
                    stream>>>(Xh, WTa, b_attn, Qb, Kb, Vb);
  flash_attn_kernel<<<dim3(TDIM / 128, NHEAD, BDIM), 256, 0, stream>>>(
      Qb, Kb, Vb, Ab);
  proj_wmma_kernel<<<dim3(CDIM / 128, (BDIM * TDIM) / 128), 256, 0, stream>>>(
      Ab, WTp, b_proj, out);
}
